// GAT3LSTM_38465727103216
// MI455X (gfx1250) — matrix operand order repaired and checked by reading, not compiled, not run on hardware
//
#include <hip/hip_runtime.h>
#include <math.h>

// Problem constants (match reference)
#define N_NODES 50000
#define N_EDGES 800000
#define NB      64
#define NH      2
#define DD      128
#define FIN     128
#define HD2     256   // NH * DD

typedef __attribute__((ext_vector_type(16))) _Float16 v16h;
typedef __attribute__((ext_vector_type(8)))  _Float16 v8h;
typedef __attribute__((ext_vector_type(8)))  float    v8f;

// ---------------------------------------------------------------- utilities

__device__ __forceinline__ void atomic_max_f32(float* addr, float val) {
    int* ai = (int*)addr;
    int old = __float_as_int(*addr);
    while (__int_as_float(old) < val) {
        int assumed = old;
        old = atomicCAS(ai, assumed, __float_as_int(val));
        if (old == assumed) break;
    }
}

__global__ void fill_f32(float* __restrict__ p, float v, int n) {
    int i = blockIdx.x * blockDim.x + threadIdx.x;
    if (i < n) p[i] = v;
}

__global__ void f32_to_f16(const float* __restrict__ x, _Float16* __restrict__ y, int n) {
    int i = blockIdx.x * blockDim.x + threadIdx.x;
    if (i < n) y[i] = (_Float16)x[i];
}

// W: [FIN, HD2] row-major f32  ->  Wt: [HD2, FIN] row-major f16
__global__ void w_transpose_f16(const float* __restrict__ W, _Float16* __restrict__ Wt) {
    int i = blockIdx.x * blockDim.x + threadIdx.x;
    if (i < FIN * HD2) {
        int k = i / HD2, n = i % HD2;
        Wt[n * FIN + k] = (_Float16)W[i];
    }
}

// ---------------------------------------------------------- WMMA FC (x @ W)
// One wave per 16x64 output tile: A fragment loaded once per K-step and
// reused across 4 WMMAs on independent accumulators (no RAW hazard chain,
// 4x less A traffic from L2 than one-tile-per-wave).
// A = xh[N,128] f16 row-major; B = wt[256,128] f16 (pre-transposed so each
// lane's fragment is contiguous halves).
// ISA 16-bit A 16x32 layout: lane = 16*g + row (g in {0,1});
//   halves 0..7  -> K = kb + 8*g + 0..7
//   halves 8..15 -> K = kb + 16 + 8*g + 0..7
// ISA 16-bit B 32x16 layout: lane = 16*g + col;
//   halves 0..7 -> K = kb + 8*g + 0..7, halves 8..15 -> K = kb + 16 + 8*g + 0..7
__global__ void __launch_bounds__(32)
fc_wmma(const _Float16* __restrict__ xh, const _Float16* __restrict__ wt,
        float* __restrict__ out) {
    const int m0   = blockIdx.x * 16;
    const int n0   = blockIdx.y * 64;
    const int lane = threadIdx.x;
    const int low  = lane & 15;
    const int g    = lane >> 4;

    const _Float16* arow = xh + (size_t)(m0 + low) * FIN;
    const _Float16* bcol = wt + (size_t)(n0 + low) * FIN;   // + j*16*FIN per subtile

    v8f acc[4];
#pragma unroll
    for (int j = 0; j < 4; ++j)
        acc[j] = (v8f){0.f, 0.f, 0.f, 0.f, 0.f, 0.f, 0.f, 0.f};

#pragma unroll
    for (int kb = 0; kb < FIN; kb += 32) {
        v8h alo = *(const v8h*)(arow + kb + g * 8);
        v8h ahi = *(const v8h*)(arow + kb + 16 + g * 8);
        v16h a;
#pragma unroll
        for (int i = 0; i < 8; ++i) { a[i] = alo[i]; a[8 + i] = ahi[i]; }

#pragma unroll
        for (int j = 0; j < 4; ++j) {
            const _Float16* bp = bcol + (size_t)j * 16 * FIN + kb + g * 8;
            v8h blo = *(const v8h*)(bp);
            v8h bhi = *(const v8h*)(bp + 16);
            v16h b;
#pragma unroll
            for (int i = 0; i < 8; ++i) { b[i] = blo[i]; b[8 + i] = bhi[i]; }
            acc[j] = __builtin_amdgcn_wmma_f32_16x16x32_f16(
                /*neg_a=*/false, a, /*neg_b=*/false, b,
                /*c_mod=*/(short)0, acc[j], /*reuse_a=*/false, /*reuse_b=*/false);
        }
    }
    // C layout: lanes 0-15 -> M = 0..7 (vgpr v), lanes 16-31 -> M = 8..15
    const int mb = g * 8;
#pragma unroll
    for (int j = 0; j < 4; ++j)
#pragma unroll
        for (int v = 0; v < 8; ++v)
            out[(size_t)(m0 + mb + v) * HD2 + n0 + j * 16 + low] = acc[j][v];
}

// ------------------------------------------------------- GAT edge machinery

// el/er: per (node, head) dot of h with al/ar. One wave per (node, head).
__global__ void __launch_bounds__(64)
node_attn(const float* __restrict__ h, const float* __restrict__ al,
          const float* __restrict__ ar, float* __restrict__ el, float* __restrict__ er) {
    int i = blockIdx.x;
    int hh = threadIdx.x >> 5;
    int lane = threadIdx.x & 31;
    const float* hv  = h  + (size_t)i * HD2 + hh * DD;
    const float* alv = al + hh * DD;
    const float* arv = ar + hh * DD;
    float sl = 0.f, sr = 0.f;
    for (int d = lane; d < DD; d += 32) {
        float x = hv[d];
        sl += x * alv[d];
        sr += x * arv[d];
    }
    for (int off = 16; off; off >>= 1) {
        sl += __shfl_down(sl, off, 32);
        sr += __shfl_down(sr, off, 32);
    }
    if (lane == 0) { el[i * NH + hh] = sl; er[i * NH + hh] = sr; }
}

// pass 1: leaky_relu score per (edge, head) + segment max over dst
__global__ void edge_score_max(const int* __restrict__ src, const int* __restrict__ dst,
                               const float* __restrict__ el, const float* __restrict__ er,
                               float* __restrict__ esc, float* __restrict__ m) {
    int t = blockIdx.x * blockDim.x + threadIdx.x;
    if (t >= N_EDGES * NH) return;
    int e = t >> 1, hh = t & 1;
    int s = src[e], d = dst[e];
    float v = el[s * NH + hh] + er[d * NH + hh];
    v = v > 0.f ? v : 0.2f * v;
    esc[t] = v;
    atomic_max_f32(&m[d * NH + hh], v);
}

// pass 2: exp(score - max) + segment sum
__global__ void edge_exp_sum(const int* __restrict__ dst, float* __restrict__ esc,
                             const float* __restrict__ m, float* __restrict__ ssum) {
    int t = blockIdx.x * blockDim.x + threadIdx.x;
    if (t >= N_EDGES * NH) return;
    int e = t >> 1, hh = t & 1;
    int d = dst[e];
    float w = __expf(esc[t] - m[d * NH + hh]);
    esc[t] = w;
    atomicAdd(&ssum[d * NH + hh], w);
}

// pass 3: agg[dst] += alpha * h[src]. One wave per (edge, head), float4 loads.
// h (51 MB) fits in the 192 MB L2 -> random gathers stay on-chip.
__global__ void __launch_bounds__(256)
edge_aggregate(const int* __restrict__ src, const int* __restrict__ dst,
               const float* __restrict__ esc, const float* __restrict__ ssum,
               const float* __restrict__ h, float* __restrict__ agg) {
    int gid  = blockIdx.x * 256 + threadIdx.x;
    int t    = gid >> 5;
    int lane = gid & 31;
    if (t >= N_EDGES * NH) return;
    int e = t >> 1, hh = t & 1;
    int s = src[e], d = dst[e];
    float alpha = esc[t] / ssum[d * NH + hh];
    float4 x = *((const float4*)(h + (size_t)s * HD2 + hh * DD) + lane);
    float* ap = agg + (size_t)d * HD2 + hh * DD + lane * 4;
    atomicAdd(ap + 0, alpha * x.x);
    atomicAdd(ap + 1, alpha * x.y);
    atomicAdd(ap + 2, alpha * x.z);
    atomicAdd(ap + 3, alpha * x.w);
}

// x_next = mean over heads of relu(agg + b)
__global__ void mean_relu(const float* __restrict__ agg, const float* __restrict__ b,
                          float* __restrict__ xout) {
    int gid = blockIdx.x * blockDim.x + threadIdx.x;
    if (gid >= N_NODES * DD) return;
    int i = gid >> 7, d = gid & 127;
    float a0 = agg[(size_t)i * HD2 + d] + b[d];
    float a1 = agg[(size_t)i * HD2 + DD + d] + b[DD + d];
    xout[gid] = 0.5f * (fmaxf(a0, 0.f) + fmaxf(a1, 0.f));
}

// ---------------------------------------------------- global attention pool

__global__ void __launch_bounds__(256)
pool_gate(const float* __restrict__ x, const float* __restrict__ gw,
          const float* __restrict__ gb, const int* __restrict__ seg,
          float* __restrict__ gate, float* __restrict__ gmax) {
    int gid  = blockIdx.x * 256 + threadIdx.x;
    int i    = gid >> 5;
    int lane = gid & 31;
    if (i >= N_NODES) return;
    const float* xv = x + (size_t)i * DD;
    float s = 0.f;
    for (int d = lane; d < DD; d += 32) s += xv[d] * gw[d];
    for (int off = 16; off; off >>= 1) s += __shfl_down(s, off, 32);
    if (lane == 0) {
        float v = s + gb[0];
        gate[i] = v;
        atomic_max_f32(&gmax[seg[i]], v);
    }
}

__global__ void pool_expsum(const int* __restrict__ seg, float* __restrict__ gate,
                            const float* __restrict__ gmax, float* __restrict__ gsum) {
    int i = blockIdx.x * blockDim.x + threadIdx.x;
    if (i >= N_NODES) return;
    float w = __expf(gate[i] - gmax[seg[i]]);
    gate[i] = w;
    atomicAdd(&gsum[seg[i]], w);
}

__global__ void __launch_bounds__(256)
pool_accum(const float* __restrict__ x, const int* __restrict__ seg,
           const float* __restrict__ gate, const float* __restrict__ gsum,
           float* __restrict__ hg) {
    int gid  = blockIdx.x * 256 + threadIdx.x;
    int i    = gid >> 5;
    int lane = gid & 31;
    if (i >= N_NODES) return;
    int sg = seg[i];
    float alpha = gate[i] / gsum[sg];
    const float* xv = x + (size_t)i * DD;
    float* o = hg + (size_t)sg * DD;
    for (int d = lane; d < DD; d += 32) atomicAdd(&o[d], alpha * xv[d]);
}

// --------------------------------------------------------------- LSTM head
// One block per batch element; full 2-layer, 4-step recurrence in LDS.
__global__ void __launch_bounds__(256)
lstm_head(const float* __restrict__ seq, const float* __restrict__ Wih,
          const float* __restrict__ Whh, const float* __restrict__ bih,
          const float* __restrict__ bhh, const float* __restrict__ cw,
          const float* __restrict__ cb, float* __restrict__ out) {
    __shared__ float xin[4][DD];
    __shared__ float hbuf[DD];
    __shared__ float cbuf[DD];
    __shared__ float gbuf[4 * DD];
    __shared__ float red[DD];
    int b = blockIdx.x, tid = threadIdx.x;

    for (int i = tid; i < 4 * DD; i += 256)
        xin[i / DD][i % DD] = seq[(i / DD) * NB * DD + b * DD + (i % DD)];
    __syncthreads();

    for (int l = 0; l < 2; ++l) {
        if (tid < DD) { hbuf[tid] = 0.f; cbuf[tid] = 0.f; }
        __syncthreads();
        for (int t = 0; t < 4; ++t) {
            for (int gidx = tid; gidx < 4 * DD; gidx += 256) {
                const float* wi = Wih + ((size_t)l * 4 * DD + gidx) * DD;
                const float* wh = Whh + ((size_t)l * 4 * DD + gidx) * DD;
                float acc = bih[l * 4 * DD + gidx] + bhh[l * 4 * DD + gidx];
                for (int k = 0; k < DD; ++k)
                    acc += xin[t][k] * wi[k] + hbuf[k] * wh[k];
                gbuf[gidx] = acc;
            }
            __syncthreads();
            if (tid < DD) {  // torch gate order i, f, g, o
                float ig = 1.f / (1.f + __expf(-gbuf[tid]));
                float fg = 1.f / (1.f + __expf(-gbuf[DD + tid]));
                float gg = tanhf(gbuf[2 * DD + tid]);
                float og = 1.f / (1.f + __expf(-gbuf[3 * DD + tid]));
                float c = fg * cbuf[tid] + ig * gg;
                cbuf[tid] = c;
                hbuf[tid] = og * tanhf(c);
            }
            __syncthreads();
            if (tid < DD) xin[t][tid] = hbuf[tid];  // input to next layer
            __syncthreads();
        }
    }
    if (tid < DD) red[tid] = hbuf[tid] * cw[tid];
    __syncthreads();
    for (int off = 64; off; off >>= 1) {
        if (tid < off) red[tid] += red[tid + off];
        __syncthreads();
    }
    if (tid == 0) out[b] = 1.f / (1.f + __expf(-(red[0] + cb[0])));
}

// ------------------------------------------------------------------- driver

extern "C" void kernel_launch(void* const* d_in, const int* in_sizes, int n_in,
                              void* d_out, int out_size, void* d_ws, size_t ws_size,
                              hipStream_t stream) {
    (void)in_sizes; (void)n_in; (void)out_size; (void)ws_size;
    const float* nfeats = (const float*)d_in[0];
    const int*   src    = (const int*)d_in[1];
    const int*   dst    = (const int*)d_in[2];
    const int*   seg    = (const int*)d_in[3];
    const float* W[3]   = {(const float*)d_in[4],  (const float*)d_in[8],  (const float*)d_in[12]};
    const float* al[3]  = {(const float*)d_in[5],  (const float*)d_in[9],  (const float*)d_in[13]};
    const float* ar[3]  = {(const float*)d_in[6],  (const float*)d_in[10], (const float*)d_in[14]};
    const float* bb[3]  = {(const float*)d_in[7],  (const float*)d_in[11], (const float*)d_in[15]};
    const float* gw[4]  = {(const float*)d_in[16], (const float*)d_in[18], (const float*)d_in[20], (const float*)d_in[22]};
    const float* gb[4]  = {(const float*)d_in[17], (const float*)d_in[19], (const float*)d_in[21], (const float*)d_in[23]};
    const float* Wih    = (const float*)d_in[24];
    const float* Whh    = (const float*)d_in[25];
    const float* bih    = (const float*)d_in[26];
    const float* bhh    = (const float*)d_in[27];
    const float* cw     = (const float*)d_in[28];
    const float* cb     = (const float*)d_in[29];
    float* out = (float*)d_out;

    // workspace carve-out
    char* ws = (char*)d_ws;
    size_t off = 0;
    auto carve = [&](size_t bytes) -> char* {
        char* p = ws + off;
        off = (off + bytes + 255) & ~(size_t)255;
        return p;
    };
    float*    xcur = (float*)carve((size_t)N_NODES * DD * 4);
    _Float16* xh   = (_Float16*)carve((size_t)N_NODES * DD * 2);
    _Float16* wt   = (_Float16*)carve((size_t)HD2 * FIN * 2);
    float*    hmat = (float*)carve((size_t)N_NODES * HD2 * 4);
    float*    el   = (float*)carve((size_t)N_NODES * NH * 4);
    float*    er   = (float*)carve((size_t)N_NODES * NH * 4);
    float*    mbuf = (float*)carve((size_t)N_NODES * NH * 4);
    float*    sbuf = (float*)carve((size_t)N_NODES * NH * 4);
    float*    esc  = (float*)carve((size_t)N_EDGES * NH * 4);
    float*    agg  = (float*)carve((size_t)N_NODES * HD2 * 4);
    float*    gate = (float*)carve((size_t)N_NODES * 4);
    float*    gmax = (float*)carve((size_t)NB * 4);
    float*    gsum = (float*)carve((size_t)NB * 4);
    float*    seqb = (float*)carve((size_t)4 * NB * DD * 4);

    auto blocks = [](int n, int bs) { return (n + bs - 1) / bs; };

    fill_f32<<<blocks(4 * NB * DD, 256), 256, 0, stream>>>(seqb, 0.f, 4 * NB * DD);

    auto pool = [&](const float* x, int t, const float* gwp, const float* gbp) {
        fill_f32<<<1, 64, 0, stream>>>(gmax, -1e30f, NB);
        fill_f32<<<1, 64, 0, stream>>>(gsum, 0.f, NB);
        pool_gate<<<blocks(N_NODES * 32, 256), 256, 0, stream>>>(x, gwp, gbp, seg, gate, gmax);
        pool_expsum<<<blocks(N_NODES, 256), 256, 0, stream>>>(seg, gate, gmax, gsum);
        pool_accum<<<blocks(N_NODES * 32, 256), 256, 0, stream>>>(x, seg, gate, gsum,
                                                                  seqb + (size_t)t * NB * DD);
    };

    auto gat = [&](const float* xin_, int li) {
        f32_to_f16<<<blocks(N_NODES * DD, 256), 256, 0, stream>>>(xin_, xh, N_NODES * DD);
        w_transpose_f16<<<blocks(FIN * HD2, 256), 256, 0, stream>>>(W[li], wt);
        dim3 g(N_NODES / 16, HD2 / 64);   // one wave -> 16x64 output tile
        fc_wmma<<<g, 32, 0, stream>>>(xh, wt, hmat);
        node_attn<<<N_NODES, 64, 0, stream>>>(hmat, al[li], ar[li], el, er);
        fill_f32<<<blocks(N_NODES * NH, 256), 256, 0, stream>>>(mbuf, -1e30f, N_NODES * NH);
        fill_f32<<<blocks(N_NODES * NH, 256), 256, 0, stream>>>(sbuf, 0.f, N_NODES * NH);
        fill_f32<<<blocks(N_NODES * HD2, 256), 256, 0, stream>>>(agg, 0.f, N_NODES * HD2);
        edge_score_max<<<blocks(N_EDGES * NH, 256), 256, 0, stream>>>(src, dst, el, er, esc, mbuf);
        edge_exp_sum<<<blocks(N_EDGES * NH, 256), 256, 0, stream>>>(dst, esc, mbuf, sbuf);
        edge_aggregate<<<blocks(N_EDGES * NH * 32, 256), 256, 0, stream>>>(src, dst, esc, sbuf,
                                                                           hmat, agg);
        mean_relu<<<blocks(N_NODES * DD, 256), 256, 0, stream>>>(agg, bb[li], xcur);
    };

    pool(nfeats, 0, gw[0], gb[0]);
    gat(nfeats, 0);
    pool(xcur, 1, gw[1], gb[1]);
    gat(xcur, 1);
    pool(xcur, 2, gw[2], gb[2]);
    gat(xcur, 2);
    pool(xcur, 3, gw[3], gb[3]);

    lstm_head<<<NB, 256, 0, stream>>>(seqb, Wih, Whh, bih, bhh, cw, cb, out);
}
